// MoELayer_5592047419817
// MI455X (gfx1250) — matrix operand order repaired and checked by reading, not compiled, not run on hardware
//
#include <hip/hip_runtime.h>

// Problem constants (from reference): B=4, S=2048 -> N=8192, D=1024, H=4096, E=8, K=2
#define N_TOK 8192
#define DIM   1024
#define HID   4096
#define NEXP  8
#define TOPK  2
#define TM    16           // tokens per expert tile (M)
#define HP    (HID + 8)    // padded LDS row stride (elements); 8208 B, 16B-aligned

typedef _Float16 v16h __attribute__((ext_vector_type(16)));
typedef float    v8f  __attribute__((ext_vector_type(8)));

union FragV { v16h v; int4 q[2]; };

static __device__ __forceinline__ float gelu_exact(float x) {
    return 0.5f * x * (1.0f + erff(x * 0.70710678118654752f));
}

// ---------------------------------------------------------------------------
// fp32 -> f16 conversion (x, W1, W2), 4 elements/thread, vectorized
// ---------------------------------------------------------------------------
__global__ void cvt_f16_kernel(const float* __restrict__ src,
                               _Float16* __restrict__ dst, size_t n) {
    size_t i = ((size_t)blockIdx.x * blockDim.x + threadIdx.x) * 4;
    if (i + 3 >= n) return;
    float4 v = *(const float4*)(src + i);
    union { _Float16 h[4]; uint2 u; } t;
    t.h[0] = (_Float16)v.x; t.h[1] = (_Float16)v.y;
    t.h[2] = (_Float16)v.z; t.h[3] = (_Float16)v.w;
    *(uint2*)(dst + i) = t.u;
}

// ---------------------------------------------------------------------------
// Gating: one wave per token. Dot products over D for E=8 logits, wave-reduce,
// top-2 softmax, emit routing lists (expert -> token slots).
// ---------------------------------------------------------------------------
__global__ __launch_bounds__(256) void gating_kernel(
    const float* __restrict__ x, const float* __restrict__ Wg,
    const float* __restrict__ bg,
    float* __restrict__ logits_out,   // [N, E]
    float* __restrict__ idx_out,      // [N, K] (indices stored as float)
    int* __restrict__ counts,         // [E]
    int* __restrict__ slot_tok,       // [E][N]
    float* __restrict__ slot_w)       // [E][N]
{
    int wave = threadIdx.x >> 5;
    int lane = threadIdx.x & 31;
    int n = blockIdx.x * 8 + wave;
    const float* xr = x + (size_t)n * DIM;

    float p[NEXP];
#pragma unroll
    for (int e = 0; e < NEXP; ++e) p[e] = 0.0f;

    for (int i = lane; i < DIM; i += 32) {
        float xv = xr[i];
        const float* wr = Wg + (size_t)i * NEXP;
#pragma unroll
        for (int e = 0; e < NEXP; ++e) p[e] += xv * wr[e];
    }
#pragma unroll
    for (int e = 0; e < NEXP; ++e) {
        for (int off = 16; off > 0; off >>= 1)
            p[e] += __shfl_xor(p[e], off, 32);
    }

    if (lane == 0) {
        float lg[NEXP];
#pragma unroll
        for (int e = 0; e < NEXP; ++e) {
            lg[e] = p[e] + bg[e];
            logits_out[(size_t)n * NEXP + e] = lg[e];
        }
        // top-2 (first occurrence wins ties, matching lax.top_k)
        int i0 = 0; float l0 = lg[0];
#pragma unroll
        for (int e = 1; e < NEXP; ++e) if (lg[e] > l0) { l0 = lg[e]; i0 = e; }
        int i1 = (i0 == 0) ? 1 : 0; float l1 = lg[i1];
#pragma unroll
        for (int e = 0; e < NEXP; ++e)
            if (e != i0 && lg[e] > l1) { l1 = lg[e]; i1 = e; }

        float ew = __expf(l1 - l0);          // <= 1
        float w0 = 1.0f / (1.0f + ew);
        float w1 = ew * w0;

        idx_out[(size_t)n * TOPK + 0] = (float)i0;
        idx_out[(size_t)n * TOPK + 1] = (float)i1;

        int p0 = atomicAdd(&counts[i0], 1);
        slot_tok[(size_t)i0 * N_TOK + p0] = n;
        slot_w  [(size_t)i0 * N_TOK + p0] = w0;
        int p1 = atomicAdd(&counts[i1], 1);
        slot_tok[(size_t)i1 * N_TOK + p1] = n;
        slot_w  [(size_t)i1 * N_TOK + p1] = w1;
    }
}

// ---------------------------------------------------------------------------
// Fused expert tile: 16 tokens, both GEMMs, h resident in LDS.
//   Phase 1: h = gelu(X[16xD] @ W1[DxH] + b1)  -> LDS (f16)
//   Phase 2: y += w * (h[16xH] @ W2[HxD] + b2) -> global atomic f32
// 4 waves; each wave owns TWO 16-wide N tiles per 128-column chunk so the A
// fragment is reused across two WMMAs (1 A b128 load / WMMA instead of 2).
// WMMA f16 layouts: A lane m = row m, K-halves {hi*8..+7, 16+hi*8..+7};
// B lane l = col n=l&15, K-halves as A, strided loads (row-major weights, no transpose).
// ---------------------------------------------------------------------------
__global__ __launch_bounds__(128) void expert_kernel(
    const _Float16* __restrict__ xh,   // [N, D] f16
    const _Float16* __restrict__ w1h,  // [E][D][H] f16 (row-major, K=D outer)
    const _Float16* __restrict__ w2h,  // [E][H][D] f16 (row-major, K=H outer)
    const float* __restrict__ b1,      // [E][H]
    const float* __restrict__ b2,      // [E][D]
    const int* __restrict__ counts,
    const int* __restrict__ slot_tok,
    const float* __restrict__ slot_w,
    float* __restrict__ y)             // [N, D] f32, pre-zeroed
{
    int e = blockIdx.y;
    int cnt = counts[e];                     // uniform
    int base = blockIdx.x * TM;
    if (base >= cnt) return;                 // uniform early-out

    extern __shared__ __align__(16) char smem[];
    _Float16* lds_h = (_Float16*)smem;       // [TM][HP]

    __shared__ int   s_tok[TM];
    __shared__ float s_w[TM];
    if (threadIdx.x < TM) {
        int idx = base + threadIdx.x;
        bool valid = idx < cnt;
        s_tok[threadIdx.x] = valid ? slot_tok[(size_t)e * N_TOK + idx] : 0;
        s_w[threadIdx.x]   = valid ? slot_w[(size_t)e * N_TOK + idx] : 0.0f;
    }
    __syncthreads();

    const int tid  = threadIdx.x;
    const int wv   = tid >> 5;
    const int lane = tid & 31;
    const int m    = lane & 15;
    const int hi   = lane >> 4;

    const _Float16* xrow = xh + (size_t)s_tok[m] * DIM;   // A row for this lane
    const _Float16* w1e  = w1h + (size_t)e * DIM * HID;
    const _Float16* w2e  = w2h + (size_t)e * HID * DIM;

    // ---------------- Phase 1: X @ W1 -> gelu -> lds_h ----------------
    for (int nc = 0; nc < HID; nc += 128) {
        int nb0 = nc + wv * 32;              // this wave's two N tiles
        int nb1 = nb0 + 16;
        float bias0 = b1[(size_t)e * HID + nb0 + m];
        float bias1 = b1[(size_t)e * HID + nb1 + m];
        const _Float16* bptr0 = w1e + nb0 + m;  // B column n=lane&15
        const _Float16* bptr1 = bptr0 + 16;
        __builtin_prefetch(bptr0 + 128, 0, 1);                   // global_prefetch_b8
        v8f acc0 = {};
        v8f acc1 = {};
#pragma unroll 2
        for (int kb = 0; kb < DIM; kb += 32) {
            FragV a, b0, b1f;
            a.q[0]   = *(const int4*)(xrow + kb + hi * 8);
            a.q[1]   = *(const int4*)(xrow + kb + 16 + hi * 8);
#pragma unroll
            for (int s = 0; s < 16; ++s) {
                int kk = (s < 8) ? (hi * 8 + s) : (16 + hi * 8 + s - 8);
                b0.v[s]  = bptr0[(size_t)kk * HID];
                b1f.v[s] = bptr1[(size_t)kk * HID];
            }
            acc0 = __builtin_amdgcn_wmma_f32_16x16x32_f16(
                false, a.v, false, b0.v, (short)0, acc0, false, false);
            acc1 = __builtin_amdgcn_wmma_f32_16x16x32_f16(
                false, a.v, false, b1f.v, (short)0, acc1, false, false);
            bptr0 += (size_t)32 * HID;
            bptr1 += (size_t)32 * HID;
        }
        // C layout: VGPR v -> row (v + 8*hi), col = lane&15
#pragma unroll
        for (int v = 0; v < 8; ++v) {
            int row = v + 8 * hi;
            lds_h[(size_t)row * HP + nb0 + m] = (_Float16)gelu_exact(acc0[v] + bias0);
            lds_h[(size_t)row * HP + nb1 + m] = (_Float16)gelu_exact(acc1[v] + bias1);
        }
    }
    __syncthreads();

    // ---------------- Phase 2: h @ W2 -> weighted atomic scatter ----------------
    const _Float16* hrow = lds_h + (size_t)m * HP;
    for (int nc = 0; nc < DIM; nc += 128) {
        int nb0 = nc + wv * 32;
        int nb1 = nb0 + 16;
        float bias0 = b2[(size_t)e * DIM + nb0 + m];
        float bias1 = b2[(size_t)e * DIM + nb1 + m];
        const _Float16* bptr0 = w2e + nb0 + m;
        const _Float16* bptr1 = bptr0 + 16;
        __builtin_prefetch(bptr0 + 128, 0, 1);
        v8f acc0 = {};
        v8f acc1 = {};
#pragma unroll 2
        for (int kb = 0; kb < HID; kb += 32) {
            FragV a, b0, b1f;
            a.q[0]   = *(const int4*)(hrow + kb + hi * 8);       // ds_load_b128
            a.q[1]   = *(const int4*)(hrow + kb + 16 + hi * 8);
#pragma unroll
            for (int s = 0; s < 16; ++s) {
                int kk = (s < 8) ? (hi * 8 + s) : (16 + hi * 8 + s - 8);
                b0.v[s]  = bptr0[(size_t)kk * DIM];
                b1f.v[s] = bptr1[(size_t)kk * DIM];
            }
            acc0 = __builtin_amdgcn_wmma_f32_16x16x32_f16(
                false, a.v, false, b0.v, (short)0, acc0, false, false);
            acc1 = __builtin_amdgcn_wmma_f32_16x16x32_f16(
                false, a.v, false, b1f.v, (short)0, acc1, false, false);
            bptr0 += (size_t)32 * DIM;
            bptr1 += (size_t)32 * DIM;
        }
#pragma unroll
        for (int v = 0; v < 8; ++v) {
            int row = v + 8 * hi;
            float o0 = s_w[row] * (acc0[v] + bias0);
            float o1 = s_w[row] * (acc1[v] + bias1);
            atomicAdd(y + (size_t)s_tok[row] * DIM + nb0 + m, o0);
            atomicAdd(y + (size_t)s_tok[row] * DIM + nb1 + m, o1);
        }
    }
}

// ---------------------------------------------------------------------------
// Launch
// ---------------------------------------------------------------------------
extern "C" void kernel_launch(void* const* d_in, const int* in_sizes, int n_in,
                              void* d_out, int out_size, void* d_ws, size_t ws_size,
                              hipStream_t stream) {
    (void)in_sizes; (void)n_in; (void)out_size; (void)ws_size;
    const float* x  = (const float*)d_in[0];
    const float* Wg = (const float*)d_in[1];
    const float* bg = (const float*)d_in[2];
    const float* W1 = (const float*)d_in[3];
    const float* b1 = (const float*)d_in[4];
    const float* W2 = (const float*)d_in[5];
    const float* b2 = (const float*)d_in[6];

    float* y          = (float*)d_out;                                   // [N, D]
    float* logits_out = (float*)d_out + (size_t)N_TOK * DIM;             // [N, E]
    float* idx_out    = logits_out + (size_t)N_TOK * NEXP;               // [N, K]

    // workspace layout
    char* p = (char*)d_ws;
    _Float16* xh  = (_Float16*)p; p += (size_t)N_TOK * DIM * 2;
    _Float16* w1h = (_Float16*)p; p += (size_t)NEXP * DIM * HID * 2;
    _Float16* w2h = (_Float16*)p; p += (size_t)NEXP * HID * DIM * 2;
    int*   counts   = (int*)p;    p += 256;
    int*   slot_tok = (int*)p;    p += (size_t)NEXP * N_TOK * 4;
    float* slot_w   = (float*)p;  p += (size_t)NEXP * N_TOK * 4;

    hipMemsetAsync(counts, 0, 256, stream);
    hipMemsetAsync(y, 0, (size_t)N_TOK * DIM * sizeof(float), stream);

    // fp32 -> f16 conversions
    {
        size_t nx = (size_t)N_TOK * DIM;
        size_t nw = (size_t)NEXP * DIM * HID;
        cvt_f16_kernel<<<(unsigned)(nx / 4 / 256), 256, 0, stream>>>(x,  xh,  nx);
        cvt_f16_kernel<<<(unsigned)(nw / 4 / 256), 256, 0, stream>>>(W1, w1h, nw);
        cvt_f16_kernel<<<(unsigned)(nw / 4 / 256), 256, 0, stream>>>(W2, w2h, nw);
    }

    gating_kernel<<<N_TOK / 8, 256, 0, stream>>>(
        x, Wg, bg, logits_out, idx_out, counts, slot_tok, slot_w);

    size_t ldsB = (size_t)TM * HP * sizeof(_Float16);   // ~131 KB
    expert_kernel<<<dim3(N_TOK / TM, NEXP), 128, ldsB, stream>>>(
        xh, w1h, w2h, b1, b2, counts, slot_tok, slot_w, y);
}
